// PolyDecoder_77111842832565
// MI455X (gfx1250) — matrix operand order repaired and checked by reading, not compiled, not run on hardware
//
#include <hip/hip_runtime.h>
#include <stdint.h>

// CDNA5 WMMA vector types
typedef __attribute__((ext_vector_type(16))) __bf16 v16bf;
typedef __attribute__((ext_vector_type(8)))  float  v8f;

#define TOTAL_TERMS 37449
#define NCH   1171          // packed K chunks: 19 (deg0-3, padded to 608) + 9*128 (feat4 * {1,z0..z7})
#define S03   616           // LDS row stride (elements) for deg0-3 block (608 used + 8 pad)
#define S4    4104          // LDS row stride (elements) for scaled feat4 (4096 used + 8 pad)
#define OFF4  (16 * S03)    // element offset of scaled buffer in LDS

static __device__ __forceinline__ unsigned short f2bf(float f) {
  union { float f; uint32_t u; } c; c.f = f;
  uint32_t u = c.u + 0x7FFFu + ((c.u >> 16) & 1u);   // round-to-nearest-even
  return (unsigned short)(u >> 16);
}

// 8 bf16 children (p * z_j, j=0..7) packed into one 16B LDS store
static __device__ __forceinline__ void store8(unsigned short* dst, float p, const float zr[8]) {
  union { unsigned short s[8]; uint4 q; } u;
#pragma unroll
  for (int j = 0; j < 8; ++j) u.s[j] = f2bf(p * zr[j]);
  *(uint4*)dst = u.q;
}

// ---------------------------------------------------------------------------
// Pack W[64,37449] f32 -> bf16 in exact WMMA B-fragment order.
// Tile (chunk c, ntile nt): 32x16 bf16 = 1KB contiguous; lane L holds
// n = L&15, K = (L>>4)*16 + h  (h = 0..15), contiguous k order, same as the A fragments in load_a.
// Packed column k -> original W column t:
//   k==0        -> t=0        (deg0)
//   8<=k<592    -> t=k-7      (deg1..deg3, LDS layout 8..591)
//   k>=608      -> t=k-23     (deg4 then deg5: 585 + (k-608))
//   else        -> zero pad
// ---------------------------------------------------------------------------
__global__ void pack_w_kernel(const float* __restrict__ W, unsigned short* __restrict__ pw) {
  const int total = NCH * 4 * 32 * 16;
  int idx = blockIdx.x * blockDim.x + threadIdx.x;
  if (idx >= total) return;
  int h  = idx & 15;
  int L  = (idx >> 4) & 31;
  int nt = (idx >> 9) & 3;
  int c  = idx >> 11;
  int o  = nt * 16 + (L & 15);
  int k  = c * 32 + (L >> 4) * 16 + h;
  float v = 0.0f;
  int t = -1;
  if (k == 0)                  t = 0;
  else if (k >= 8 && k < 592)  t = k - 7;
  else if (k >= 608)           t = k - 23;
  if (t >= 0) v = W[(size_t)o * TOTAL_TERMS + t];
  pw[idx] = f2bf(v);
}

// A fragment (16x32 bf16): lane m=L&15, khi=L>>4;
// halves 0-15 = K(kbase + khi*16 ..+15), matching packed B order
static __device__ __forceinline__ v16bf load_a(const unsigned short* row, int kbase, int khi) {
  union { uint4 q[2]; v16bf v; } u;
  u.q[0] = *(const uint4*)(row + kbase + khi * 16);
  u.q[1] = *(const uint4*)(row + kbase + khi * 16 + 8);
  return u.v;
}

static __device__ __forceinline__ v16bf load_b(const unsigned short* pw, int c, int nt, int lane) {
  union { uint4 q[2]; v16bf v; } u;
  const uint4* p = (const uint4*)(pw + (((size_t)c * 4 + nt) * 32 + lane) * 16);
  u.q[0] = p[0];
  u.q[1] = p[1];
  return u.v;
}

// ---------------------------------------------------------------------------
// One block = 16 batch rows, 128 threads = 4 waves (one 16-col N-tile each).
// LDS: feats03[16][S03] (deg0..3) + scaled[16][S4] (feat4 * scale, reused 9x).
// ---------------------------------------------------------------------------
__global__ void __launch_bounds__(128) poly_gemm_kernel(
    const float* __restrict__ z, const unsigned short* __restrict__ pw,
    const float* __restrict__ bias_p, float* __restrict__ out) {
  extern __shared__ __align__(16) unsigned short lds[];

  const int tid   = threadIdx.x;
  const int r     = tid & 15;   // batch row within tile (build phase)
  const int seg   = tid >> 4;   // 0..7 work segment (build phase)
  const int lane  = tid & 31;
  const int wave  = tid >> 5;   // = N tile index 0..3
  const int mbase = blockIdx.x * 16;

  float zr[8];
#pragma unroll
  for (int j = 0; j < 8; ++j) zr[j] = z[(size_t)(mbase + r) * 8 + j];

  // ---- build deg0..3 features (each thread handles (row r, segment seg)) ----
  unsigned short* f03 = lds + r * S03;
  if (seg == 0) {
    union { unsigned short s[8]; uint4 q; } u;
    u.s[0] = f2bf(1.0f);
#pragma unroll
    for (int j = 1; j < 8; ++j) u.s[j] = 0;
    *(uint4*)(f03 + 0) = u.q;                    // deg0 + zero pad
    store8(f03 + 8, 1.0f, zr);                   // deg1: z_j
  }
  if (seg == 1 || seg == 2) {                    // zero tail 592..607
    uint4 zq = {0, 0, 0, 0};
    *(uint4*)(f03 + 592 + (seg - 1) * 8) = zq;
  }
  store8(f03 + 16 + seg * 8, zr[seg], zr);       // deg2 block i=seg
#pragma unroll
  for (int t2 = seg * 8; t2 < seg * 8 + 8; ++t2) // deg3 parents
    store8(f03 + 80 + t2 * 8, zr[t2 >> 3] * zr[t2 & 7], zr);

  // ---- build scaled feat4 buffer for pass s (scale = 1 or z_{s-1}) ----
  unsigned short* f4 = lds + OFF4 + r * S4;
  auto build_scaled = [&](int s) {
    float sc = (s == 0) ? 1.0f : zr[s - 1];
    for (int t3l = 0; t3l < 64; ++t3l) {
      int t3 = seg * 64 + t3l;
      float p = sc * zr[(t3 >> 6) & 7] * zr[(t3 >> 3) & 7] * zr[t3 & 7];
      store8(f4 + t3 * 8, p, zr);
    }
  };
  build_scaled(0);

  // ---- GEMM: wave `wave` owns output columns [wave*16, wave*16+16) ----
  const int n   = lane & 15;
  const int khi = lane >> 4;
  const unsigned short* arow03 = lds + n * 0 + (lane & 15) * S03;  // m = lane&15
  const unsigned short* arow4  = lds + OFF4 + (lane & 15) * S4;

  float bias = bias_p[wave * 16 + n];
  v8f acc0, acc1;
#pragma unroll
  for (int v = 0; v < 8; ++v) { acc0[v] = bias; acc1[v] = 0.0f; }

  __syncthreads();

  // chunks 0..18: deg0..3 columns
  for (int c = 0; c < 19; c += 2) {
    v16bf a0 = load_a(arow03, c * 32, khi);
    v16bf b0 = load_b(pw, c, wave, lane);
    acc0 = __builtin_amdgcn_wmma_f32_16x16x32_bf16(false, a0, false, b0, (short)0, acc0, false, false);
    if (c + 1 < 19) {
      v16bf a1 = load_a(arow03, (c + 1) * 32, khi);
      v16bf b1 = load_b(pw, c + 1, wave, lane);
      acc1 = __builtin_amdgcn_wmma_f32_16x16x32_bf16(false, a1, false, b1, (short)0, acc1, false, false);
    }
  }

  // 9 passes over scaled feat4: s=0 -> deg4 columns, s=1..8 -> deg5 (i1 = s-1)
  for (int s = 0; s < 9; ++s) {
    if (s > 0) {
      __syncthreads();          // all waves finished reading scaled(s-1)
      build_scaled(s);
      __syncthreads();
    }
    const int cbase = 19 + s * 128;
#pragma unroll 4
    for (int cc = 0; cc < 128; cc += 2) {
      v16bf a0 = load_a(arow4, cc * 32, khi);
      v16bf b0 = load_b(pw, cbase + cc, wave, lane);
      acc0 = __builtin_amdgcn_wmma_f32_16x16x32_bf16(false, a0, false, b0, (short)0, acc0, false, false);
      v16bf a1 = load_a(arow4, (cc + 1) * 32, khi);
      v16bf b1 = load_b(pw, cbase + cc + 1, wave, lane);
      acc1 = __builtin_amdgcn_wmma_f32_16x16x32_bf16(false, a1, false, b1, (short)0, acc1, false, false);
    }
  }

  // store: C/D layout (ISA 7.12.2): lane holds n=lane&15, VGPR v -> M = v + 8*khi
#pragma unroll
  for (int v = 0; v < 8; ++v) {
    int mm = v + 8 * khi;
    out[(size_t)(mbase + mm) * 64 + wave * 16 + n] = acc0[v] + acc1[v];
  }
}

extern "C" void kernel_launch(void* const* d_in, const int* in_sizes, int n_in,
                              void* d_out, int out_size, void* d_ws, size_t ws_size,
                              hipStream_t stream) {
  const float* z = (const float*)d_in[0];   // [4096, 8]
  const float* W = (const float*)d_in[1];   // [64, 37449]
  const float* b = (const float*)d_in[2];   // [64]
  float* out = (float*)d_out;               // [4096, 64]
  unsigned short* pw = (unsigned short*)d_ws;  // 4.8 MB packed bf16 W

  const int total_pack = NCH * 4 * 32 * 16;
  hipLaunchKernelGGL(pack_w_kernel, dim3((total_pack + 255) / 256), dim3(256), 0, stream,
                     W, pw);

  const size_t lds_bytes = (size_t)(16 * S03 + 16 * S4) * sizeof(unsigned short); // ~151 KB
  hipLaunchKernelGGL(poly_gemm_kernel, dim3(4096 / 16), dim3(128), lds_bytes, stream,
                     z, pw, b, out);
}
